// LSTMModel_31172872634925
// MI455X (gfx1250) — matrix operand order repaired and checked by reading, not compiled, not run on hardware
//
#include <hip/hip_runtime.h>
#include <hip/hip_bf16.h>
#include <stdint.h>

// Problem constants (match reference)
#define BB   128
#define TT   1024
#define DD   48
#define HH   256
#define H4   1024   // 4*H
#define DP   64     // D padded to K-tile multiple
#define X1W  512    // 2*H (layer-1 input width)
#define KP0  320    // layer0 Kpad = 256 (h) + 64 (x pad)
#define KP1  768    // layer1 Kpad = 256 (h) + 512 (y0)

typedef __attribute__((ext_vector_type(16))) __bf16   v16bf;
typedef __attribute__((ext_vector_type(8)))  float    v8f;
typedef __attribute__((ext_vector_type(4)))  unsigned u32x4;

union ABu { v16bf v; u32x4 q[2]; };

__device__ __forceinline__ unsigned short f2bf(float f) {
  unsigned u = __float_as_uint(f);
  u += 0x7FFFu + ((u >> 16) & 1u);          // round-to-nearest-even
  return (unsigned short)(u >> 16);
}
__device__ __forceinline__ float sigmoid_fast(float x) {
  return 1.0f / (1.0f + __expf(-x));
}
__device__ __forceinline__ float tanh_fast(float x) {
  return 1.0f - 2.0f / (__expf(2.0f * x) + 1.0f);
}

// ---------------------------------------------------------------------------
// Repack one direction's LSTM weights into bf16 Wcat[4H][Kpad], K = [h | x],
// and fold the two biases into one f32 bias[4H].
// ---------------------------------------------------------------------------
__global__ void pack_w_kernel(const float* __restrict__ wih,
                              const float* __restrict__ whh,
                              const float* __restrict__ bih,
                              const float* __restrict__ bhh,
                              unsigned short* __restrict__ wcat,
                              float* __restrict__ bias,
                              int F, int Kpad) {
  long idx = (long)blockIdx.x * blockDim.x + threadIdx.x;
  long total = (long)H4 * Kpad;
  if (idx >= total) return;
  int col = (int)(idx / Kpad);
  int k   = (int)(idx % Kpad);
  float v;
  if (k < HH)            v = whh[(long)col * HH + k];
  else if (k - HH < F)   v = wih[(long)col * F + (k - HH)];
  else                   v = 0.0f;
  wcat[idx] = f2bf(v);
  if (k == 0) bias[col] = bih[col] + bhh[col];
}

// x [B,T,D] f32  ->  xpad [T][B][DP] bf16 (zero padded D->DP)
__global__ void pack_x_kernel(const float* __restrict__ x,
                              unsigned short* __restrict__ xp) {
  long idx = (long)blockIdx.x * blockDim.x + threadIdx.x;
  long total = (long)TT * BB * DP;
  if (idx >= total) return;
  int d = (int)(idx % DP);
  long tb = idx / DP;
  int b = (int)(tb % BB);
  int t = (int)(tb / BB);
  float v = (d < DD) ? x[((long)b * TT + t) * DD + d] : 0.0f;
  xp[idx] = f2bf(v);
}

// Per-direction grid barrier across the 16 blocks owning that direction.
__device__ __forceinline__ void grid_sync_16(unsigned* bar, unsigned& phase) {
  __syncthreads();
  if (threadIdx.x == 0) {
    __threadfence();
    atomicAdd(bar, 1u);
    const unsigned target = (phase + 1u) * 16u;
    volatile unsigned* vb = (volatile unsigned*)bar;
    while (*vb < target) __builtin_amdgcn_s_sleep(2);
    __threadfence();
  }
  __syncthreads();
  ++phase;
}

// ---------------------------------------------------------------------------
// Persistent recurrent kernel: one launch runs a full bidirectional layer.
// grid = 32 blocks (dir = blockIdx>>4, jblock = blockIdx&15), block = 256.
// Each workgroup: 16 hidden columns (=> 64 gate columns, weights in LDS).
// Each wave: one 16-row block of the batch; c/h state kept in v8f registers.
// ---------------------------------------------------------------------------
template <int KPAD, int NKEXT>
__global__ __launch_bounds__(256)
void lstm_layer_kernel(const unsigned short* __restrict__ wcatF,
                       const unsigned short* __restrict__ wcatB,
                       const float* __restrict__ biasF,
                       const float* __restrict__ biasB,
                       const unsigned short* __restrict__ xext,  // per-t ext features
                       long  xextTStride,                        // elements per t
                       int   extStride,                          // elements per row
                       const int* __restrict__ lengths,
                       unsigned short* __restrict__ hbuf,        // [dir][2][B][H] bf16
                       unsigned short* __restrict__ yout,        // [T][B][2H] or null
                       float* __restrict__ hfinal,               // [B][2H] or null
                       unsigned* __restrict__ barBase) {
  constexpr int NK = 8 + NKEXT;     // K tiles of 32: 8 for h, NKEXT for ext
  const int tid  = threadIdx.x;
  const int lane = tid & 31;
  const int wave = tid >> 5;                 // 0..7 -> row block
  const int dir  = blockIdx.x >> 4;          // 0 = fwd, 1 = bwd
  const int jblk = blockIdx.x & 15;
  const int j0   = jblk * 16;
  const int rowBase = wave * 16;

  const unsigned short* wcat = dir ? wcatB : wcatF;
  const float*          bias = dir ? biasB : biasF;
  unsigned* bar = barBase + dir * 64;        // separate cacheline per direction

  // ---- stage this WG's 64 weight columns into LDS (ds_store_b128 path) ----
  extern __shared__ unsigned short ldsw[];   // [64][KPAD] bf16
  {
    const int vecPerGate = (16 * KPAD) / 8;  // u32x4 = 8 bf16
    for (int g = 0; g < 4; ++g) {
      const u32x4* src = (const u32x4*)(wcat + (size_t)(g * HH + j0) * KPAD);
      u32x4* dst = (u32x4*)(ldsw + (size_t)g * 16 * KPAD);
      for (int i = tid; i < vecPerGate; i += 256) dst[i] = src[i];
    }
  }
  __syncthreads();

  // ---- per-lane geometry (matches CDNA5 16-bit WMMA VGPR layouts) ----
  const int lm   = lane & 15;     // A/B/D: M-or-N index within tile
  const int hi   = lane >> 4;     // lane half
  const int jcol = j0 + lm;       // owned hidden column
  const int aKb  = hi * 8;        // A: K sub-offset for this half
  const int bKh  = hi * 8;        // B: K sub-offset for this half

  const float bI = bias[0 * HH + jcol];
  const float bF = bias[1 * HH + jcol];
  const float bG = bias[2 * HH + jcol];
  const float bO = bias[3 * HH + jcol];

  int lenv[8];
#pragma unroll
  for (int r = 0; r < 8; ++r) lenv[r] = lengths[rowBase + hi * 8 + r];

  v8f c = {0.f, 0.f, 0.f, 0.f, 0.f, 0.f, 0.f, 0.f};
  v8f h = {0.f, 0.f, 0.f, 0.f, 0.f, 0.f, 0.f, 0.f};

  unsigned short* hb0 = hbuf + (size_t)dir * 2 * BB * HH;   // ping-pong base
  int p = 0;
  unsigned phase = 0;

  for (int s = 0; s < TT; ++s) {
    const int t = dir ? (TT - 1 - s) : s;
    const unsigned short* hsrc = hb0 + (size_t)p * BB * HH;
    const unsigned short* xsrc = xext + (size_t)t * xextTStride;

    // prefetch next step's external input row (global_prefetch_b8)
    if (s + 1 < TT) {
      const int tn = dir ? (t - 1) : (t + 1);
      __builtin_prefetch(xext + (size_t)tn * xextTStride +
                         (size_t)(rowBase + lm) * extStride, 0, 1);
    }

    v8f acc[4];
#pragma unroll
    for (int r = 0; r < 8; ++r) {
      acc[0][r] = bI; acc[1][r] = bF; acc[2][r] = bG; acc[3][r] = bO;
    }

#pragma unroll
    for (int ks = 0; ks < NK; ++ks) {
      const unsigned short* abase;
      int astr, kr;
      if (ks < 8) { abase = hsrc; astr = HH;        kr = ks * 32; }
      else        { abase = xsrc; astr = extStride; kr = (ks - 8) * 32; }

      // A fragment: row = batch row, two 16B loads cover K{kb..kb+7, kb+16..}
      ABu A;
      const unsigned short* ap =
          abase + (size_t)(rowBase + lm) * astr + kr + aKb;
      A.q[0] = *(const u32x4*)ap;
      A.q[1] = *(const u32x4*)(ap + 16);

      const int k0 = ks * 32;
#pragma unroll
      for (int g = 0; g < 4; ++g) {
        // B fragment from LDS: K{kb..kb+7, kb+16..kb+23} for column (g*16+lm)
        ABu Bt;
        const unsigned short* bp =
            ldsw + (size_t)(g * 16 + lm) * KPAD + k0 + bKh;
        Bt.q[0] = *(const u32x4*)bp;
        Bt.q[1] = *(const u32x4*)(bp + 16);
        acc[g] = __builtin_amdgcn_wmma_f32_16x16x32_bf16(
            false, A.v, false, Bt.v, (short)0, acc[g], false, false);
      }
    }

    // ---- gate activations + masked state update (c/h in registers) ----
    unsigned short* hdst = hb0 + (size_t)(1 - p) * BB * HH;
#pragma unroll
    for (int r = 0; r < 8; ++r) {
      const int row = rowBase + hi * 8 + r;
      const bool valid = t < lenv[r];
      const float iv = sigmoid_fast(acc[0][r]);
      const float fv = sigmoid_fast(acc[1][r]);
      const float gv = tanh_fast(acc[2][r]);
      const float ov = sigmoid_fast(acc[3][r]);
      const float cn = fv * c[r] + iv * gv;
      const float hn = ov * tanh_fast(cn);
      if (valid) { c[r] = cn; h[r] = hn; }
      hdst[(size_t)row * HH + jcol] = f2bf(h[r]);   // carried state
      if (yout)                                     // zeroed padded outputs
        yout[(size_t)t * BB * X1W + (size_t)row * X1W + dir * HH + jcol] =
            f2bf(valid ? hn : 0.0f);
    }

    grid_sync_16(bar, phase);
    p ^= 1;
  }

  if (hfinal) {
#pragma unroll
    for (int r = 0; r < 8; ++r) {
      const int row = rowBase + hi * 8 + r;
      hfinal[(size_t)row * X1W + dir * HH + jcol] = h[r];
    }
  }
}

// ---------------------------------------------------------------------------
// FC head: out[m] = fc2_b + relu(hidden[m]@fc1_w^T + fc1_b) @ fc2_w^T
// ---------------------------------------------------------------------------
__global__ void fc_head_kernel(const float* __restrict__ hidden,
                               const float* __restrict__ fc1w,
                               const float* __restrict__ fc1b,
                               const float* __restrict__ fc2w,
                               const float* __restrict__ fc2b,
                               float* __restrict__ out) {
  __shared__ float sh[256];
  const int m = blockIdx.x;
  const int n = threadIdx.x;
  const float* hrow = hidden + (size_t)m * X1W;
  const float* w = fc1w + (size_t)n * X1W;
  float s = fc1b[n];
  for (int k = 0; k < X1W; ++k) s += hrow[k] * w[k];
  s = fmaxf(s, 0.0f);
  sh[n] = s * fc2w[n];
  __syncthreads();
  for (int off = 128; off > 0; off >>= 1) {
    if (n < off) sh[n] += sh[n + off];
    __syncthreads();
  }
  if (n == 0) out[m] = sh[0] + fc2b[0];
}

// ---------------------------------------------------------------------------
extern "C" void kernel_launch(void* const* d_in, const int* in_sizes, int n_in,
                              void* d_out, int out_size, void* d_ws, size_t ws_size,
                              hipStream_t stream) {
  (void)in_sizes; (void)n_in; (void)out_size; (void)ws_size;

  const float* x        = (const float*)d_in[0];
  const int*   lengths  = (const int*)d_in[1];
  const float* w_ih_l0f = (const float*)d_in[2];
  const float* w_hh_l0f = (const float*)d_in[3];
  const float* b_ih_l0f = (const float*)d_in[4];
  const float* b_hh_l0f = (const float*)d_in[5];
  const float* w_ih_l0b = (const float*)d_in[6];
  const float* w_hh_l0b = (const float*)d_in[7];
  const float* b_ih_l0b = (const float*)d_in[8];
  const float* b_hh_l0b = (const float*)d_in[9];
  const float* w_ih_l1f = (const float*)d_in[10];
  const float* w_hh_l1f = (const float*)d_in[11];
  const float* b_ih_l1f = (const float*)d_in[12];
  const float* b_hh_l1f = (const float*)d_in[13];
  const float* w_ih_l1b = (const float*)d_in[14];
  const float* w_hh_l1b = (const float*)d_in[15];
  const float* b_ih_l1b = (const float*)d_in[16];
  const float* b_hh_l1b = (const float*)d_in[17];
  const float* fc1w     = (const float*)d_in[18];
  const float* fc1b     = (const float*)d_in[19];
  const float* fc2w     = (const float*)d_in[20];
  const float* fc2b     = (const float*)d_in[21];
  float* out = (float*)d_out;

  // ---- workspace carve-up ----
  char* ws = (char*)d_ws;
  size_t off = 0;
  unsigned* bar = (unsigned*)(ws + off);              off += 1024;                      // 4 dir/layer counters, cacheline apart
  float* bias = (float*)(ws + off);                   off += (size_t)4 * H4 * 4;
  unsigned short* wcL0F = (unsigned short*)(ws + off); off += (size_t)H4 * KP0 * 2;
  unsigned short* wcL0B = (unsigned short*)(ws + off); off += (size_t)H4 * KP0 * 2;
  unsigned short* wcL1F = (unsigned short*)(ws + off); off += (size_t)H4 * KP1 * 2;
  unsigned short* wcL1B = (unsigned short*)(ws + off); off += (size_t)H4 * KP1 * 2;
  unsigned short* xpad  = (unsigned short*)(ws + off); off += (size_t)TT * BB * DP * 2;
  unsigned short* y0    = (unsigned short*)(ws + off); off += (size_t)TT * BB * X1W * 2;
  unsigned short* hbuf0 = (unsigned short*)(ws + off); off += (size_t)2 * 2 * BB * HH * 2;
  unsigned short* hbuf1 = (unsigned short*)(ws + off); off += (size_t)2 * 2 * BB * HH * 2;
  float* hidden = (float*)(ws + off);                  off += (size_t)BB * X1W * 4;

  // zero the state that must start at zero (barriers, h ping-pong buffers)
  hipMemsetAsync(bar, 0, 1024, stream);
  hipMemsetAsync(hbuf0, 0, (size_t)2 * 2 * BB * HH * 2, stream);
  hipMemsetAsync(hbuf1, 0, (size_t)2 * 2 * BB * HH * 2, stream);

  // ---- weight/bias/x repack ----
  {
    long tot0 = (long)H4 * KP0;
    long tot1 = (long)H4 * KP1;
    pack_w_kernel<<<(tot0 + 255) / 256, 256, 0, stream>>>(
        w_ih_l0f, w_hh_l0f, b_ih_l0f, b_hh_l0f, wcL0F, bias + 0 * H4, DD, KP0);
    pack_w_kernel<<<(tot0 + 255) / 256, 256, 0, stream>>>(
        w_ih_l0b, w_hh_l0b, b_ih_l0b, b_hh_l0b, wcL0B, bias + 1 * H4, DD, KP0);
    pack_w_kernel<<<(tot1 + 255) / 256, 256, 0, stream>>>(
        w_ih_l1f, w_hh_l1f, b_ih_l1f, b_hh_l1f, wcL1F, bias + 2 * H4, X1W, KP1);
    pack_w_kernel<<<(tot1 + 255) / 256, 256, 0, stream>>>(
        w_ih_l1b, w_hh_l1b, b_ih_l1b, b_hh_l1b, wcL1B, bias + 3 * H4, X1W, KP1);
    long totx = (long)TT * BB * DP;
    pack_x_kernel<<<(totx + 255) / 256, 256, 0, stream>>>(x, xpad);
  }

  // ---- layer 0 (both directions concurrently), outputs -> y0 ----
  const int smem0 = 64 * KP0 * 2;   // 40 KB weight stage
  const int smem1 = 64 * KP1 * 2;   // 96 KB weight stage (WGP has 320 KB)
  hipFuncSetAttribute(reinterpret_cast<const void*>(&lstm_layer_kernel<KP0, 2>),
                      hipFuncAttributeMaxDynamicSharedMemorySize, smem0);
  hipFuncSetAttribute(reinterpret_cast<const void*>(&lstm_layer_kernel<KP1, 16>),
                      hipFuncAttributeMaxDynamicSharedMemorySize, smem1);

  lstm_layer_kernel<KP0, 2><<<32, 256, smem0, stream>>>(
      wcL0F, wcL0B, bias + 0 * H4, bias + 1 * H4,
      xpad, (long)BB * DP, DP, lengths,
      hbuf0, y0, nullptr, bar + 0);

  // ---- layer 1, consumes y0, produces final hidden state ----
  lstm_layer_kernel<KP1, 16><<<32, 256, smem1, stream>>>(
      wcL1F, wcL1B, bias + 2 * H4, bias + 3 * H4,
      y0, (long)BB * X1W, X1W, lengths,
      hbuf1, nullptr, hidden, bar + 128);

  // ---- FC head ----
  fc_head_kernel<<<BB, 256, 0, stream>>>(hidden, fc1w, fc1b, fc2w, fc2b, out);
}
